// MillionsMoE_523986010200
// MI455X (gfx1250) — matrix operand order repaired and checked by reading, not compiled, not run on hardware
//
#include <hip/hip_runtime.h>
#include <hip/hip_bf16.h>
#include <stdint.h>

typedef __bf16 bf16;
typedef __attribute__((ext_vector_type(16))) __bf16 v16bf;
typedef __attribute__((ext_vector_type(4)))  __bf16 v4bf;
typedef __attribute__((ext_vector_type(8)))  float  v8f;

constexpr int D     = 512;
constexpr int H     = 8;
constexpr int DK    = 128;
constexpr int HALF  = 64;
constexpr int E     = 256;
constexpr int K     = 16;
constexpr int NT    = 1024;         // N*T
constexpr int HD    = H * DK;       // 1024
constexpr int MROWS = NT * H;       // 8192
constexpr int SLOTS = MROWS * K;    // 131072
constexpr int EE    = E * E;        // 65536 expert rows per table

// ---------------------------------------------------------------------------
// WMMA fragment loaders (bf16, 16x16x32), A/B in contiguous k order k = 16h + s (same for both).
// A (16x32 MxK): lane L: M=L%16, h=L/16; VGPR v: K={16h+2v, 16h+2v+1}
// B (32x16 KxN): lane L: N=L%16, h=L/16; VGPR v: K={16h+2v, 16h+2v+1}
// C/D (16x16 f32): VGPR r, lane L: N=L%16, M=r+8*(L/16)
// ---------------------------------------------------------------------------
__device__ __forceinline__ v16bf load_frag_a(const bf16* __restrict__ A, int lda, int lane) {
  int m = lane & 15, h = lane >> 4;
  const bf16* row = A + m * lda;
  v16bf a;
#pragma unroll
  for (int v = 0; v < 8; ++v) {
    int k = 16 * h + 2 * v;
    a[2 * v]     = row[k];
    a[2 * v + 1] = row[k + 1];
  }
  return a;
}

// B stored row-major K x N (leading dim ldb): element (k,n) = B[k*ldb + n]
__device__ __forceinline__ v16bf load_frag_b(const bf16* __restrict__ B, int ldb, int lane) {
  int n = lane & 15, h = lane >> 4;
  v16bf b;
#pragma unroll
  for (int v = 0; v < 8; ++v) {
    int k = 16 * h + 2 * v;
    b[2 * v]     = B[k * ldb + n];
    b[2 * v + 1] = B[(k + 1) * ldb + n];
  }
  return b;
}

// B given transposed (N-major rows of length K): element (k,n) = Bt[n*ldt + k]
__device__ __forceinline__ v16bf load_frag_bt(const bf16* __restrict__ Bt, int ldt, int lane) {
  int n = lane & 15, h = lane >> 4;
  const bf16* row = Bt + n * ldt;
  v16bf b;
#pragma unroll
  for (int v = 0; v < 8; ++v) {
    int k = 16 * h + 2 * v;
    b[2 * v]     = row[k];
    b[2 * v + 1] = row[k + 1];
  }
  return b;
}

__device__ __forceinline__ v8f wmma_bf16(v16bf a, v16bf b, v8f c) {
  return __builtin_amdgcn_wmma_f32_16x16x32_bf16(false, a, false, b, (short)0, c, false, false);
}

// ---------------------------------------------------------------------------
// CDNA5 async LDS copy path (ASYNCcnt-tracked, bypasses VGPRs)
// ---------------------------------------------------------------------------
__device__ __forceinline__ void async_copy_b128(uint32_t lds_addr, const void* gaddr) {
  asm volatile("global_load_async_to_lds_b128 %0, %1, off"
               :: "v"(lds_addr), "v"((uint64_t)(uintptr_t)gaddr)
               : "memory");
}

__device__ __forceinline__ void wait_async0() {
#if __has_builtin(__builtin_amdgcn_s_wait_asynccnt)
  __builtin_amdgcn_s_wait_asynccnt(0);
#else
  asm volatile("s_wait_asynccnt 0x0" ::: "memory");
#endif
}

// ---------------------------------------------------------------------------
// Utility kernels
// ---------------------------------------------------------------------------
__global__ void cvt_bf16_kernel(const float* __restrict__ src, bf16* __restrict__ dst, int n) {
  int i = blockIdx.x * blockDim.x + threadIdx.x;
  if (i < n) dst[i] = (bf16)src[i];
}

// vectorized table conversion: 4 floats -> packed 4 bf16 (8B store)
__global__ void cvt_tables4_kernel(const float* __restrict__ src, bf16* __restrict__ dst, int n4) {
  int i = blockIdx.x * blockDim.x + threadIdx.x;
  if (i < n4) {
    float4 f = ((const float4*)src)[i];
    v4bf o = {(bf16)f.x, (bf16)f.y, (bf16)f.z, (bf16)f.w};
    *((v4bf*)dst + i) = o;
  }
}

__global__ void zero_out_kernel(float* __restrict__ out, int n) {
  int i = blockIdx.x * blockDim.x + threadIdx.x;
  if (i < n) out[i] = 0.f;
}

// ---------------------------------------------------------------------------
// qh = Q (NTxD) * Wq^T (HDxD) + bq  -> (NT x HD) f32
// ---------------------------------------------------------------------------
__global__ __launch_bounds__(256)
void qproj_kernel(const bf16* __restrict__ Qb, const bf16* __restrict__ Wqb,
                  const float* __restrict__ bq, float* __restrict__ qh) {
  int lane = threadIdx.x & 31;
  int wave = threadIdx.x >> 5;
  int tile = blockIdx.x * 8 + wave;     // 64 x 64 tiles
  int tm = tile & 63;
  int tn = tile >> 6;
  v8f acc = {};
  const bf16* Abase = Qb  + (tm * 16) * D;
  const bf16* Bbase = Wqb + (tn * 16) * D;
#pragma unroll
  for (int ks = 0; ks < D / 32; ++ks) {
    v16bf a = load_frag_a (Abase + ks * 32, D, lane);
    v16bf b = load_frag_bt(Bbase + ks * 32, D, lane);
    acc = wmma_bf16(a, b, acc);
  }
  int n = lane & 15, h = lane >> 4;
  int col  = tn * 16 + n;
  float bias = bq[col];
#pragma unroll
  for (int r = 0; r < 8; ++r) {
    int row = tm * 16 + r + 8 * h;
    qh[row * HD + col] = acc[r] + bias;
  }
}

// ---------------------------------------------------------------------------
// s1/s2[(t*H+h)*E + e] = dot(qh[t, h*DK + side*HALF .. +64], keys[h,side,e,:])
// ---------------------------------------------------------------------------
__global__ __launch_bounds__(256)
void scores_kernel(const float* __restrict__ qh, const float* __restrict__ keys,
                   float* __restrict__ s1, float* __restrict__ s2) {
  int gid  = blockIdx.x * blockDim.x + threadIdx.x;  // 2 * 8192 * 256 threads
  int side = gid >> 21;
  int rem  = gid & ((1 << 21) - 1);
  int row  = rem >> 8;        // 0..8191
  int e    = rem & 255;
  int t = row >> 3, h = row & 7;
  const float4* qv = (const float4*)(qh + t * HD + h * DK + side * HALF);
  const float4* kv = (const float4*)(keys + ((size_t)((h * 2 + side) * E + e)) * HALF);
  float acc = 0.f;
#pragma unroll
  for (int i = 0; i < HALF / 4; ++i) {
    float4 q = qv[i], k = kv[i];
    acc += q.x * k.x + q.y * k.y + q.z * k.z + q.w * k.w;
  }
  (side ? s2 : s1)[row * E + e] = acc;
}

// ---------------------------------------------------------------------------
// top-16 of each side, combine 16x16, top-16 of combos, softmax
// ---------------------------------------------------------------------------
__global__ __launch_bounds__(128)
void topk_kernel(const float* __restrict__ s1g, const float* __restrict__ s2g,
                 int* __restrict__ idxg, float* __restrict__ wgtg) {
  __shared__ float l1v[4][16]; __shared__ int l1i[4][16];
  __shared__ float l2v[4][16]; __shared__ int l2i[4][16];
  int lane = threadIdx.x & 31;
  int w    = threadIdx.x >> 5;
  int row  = blockIdx.x * 4 + w;
  const float* S1 = s1g + row * E;
  const float* S2 = s2g + row * E;
  float v1[8], v2[8];
#pragma unroll
  for (int e = 0; e < 8; ++e) { v1[e] = S1[lane * 8 + e]; v2[e] = S2[lane * 8 + e]; }

#pragma unroll
  for (int j = 0; j < 16; ++j) {          // side 1
    float lm = -1e30f; int li = 0;
#pragma unroll
    for (int e = 0; e < 8; ++e) if (v1[e] > lm) { lm = v1[e]; li = lane * 8 + e; }
#pragma unroll
    for (int off = 16; off > 0; off >>= 1) {
      float ov = __shfl_xor(lm, off, 32); int oi = __shfl_xor(li, off, 32);
      if (ov > lm || (ov == lm && oi < li)) { lm = ov; li = oi; }
    }
    if (lane == 0) { l1v[w][j] = lm; l1i[w][j] = li; }
    if ((li >> 3) == lane) {
#pragma unroll
      for (int e = 0; e < 8; ++e) if (e == (li & 7)) v1[e] = -1e30f;
    }
  }
#pragma unroll
  for (int j = 0; j < 16; ++j) {          // side 2
    float lm = -1e30f; int li = 0;
#pragma unroll
    for (int e = 0; e < 8; ++e) if (v2[e] > lm) { lm = v2[e]; li = lane * 8 + e; }
#pragma unroll
    for (int off = 16; off > 0; off >>= 1) {
      float ov = __shfl_xor(lm, off, 32); int oi = __shfl_xor(li, off, 32);
      if (ov > lm || (ov == lm && oi < li)) { lm = ov; li = oi; }
    }
    if (lane == 0) { l2v[w][j] = lm; l2i[w][j] = li; }
    if ((li >> 3) == lane) {
#pragma unroll
      for (int e = 0; e < 8; ++e) if (e == (li & 7)) v2[e] = -1e30f;
    }
  }
  __syncthreads();

  float pv[8]; int pp[8];
#pragma unroll
  for (int e = 0; e < 8; ++e) {
    int c = lane * 8 + e, p = c >> 4, q = c & 15;
    pv[e] = l1v[w][p] + l2v[w][q];
    pp[e] = l1i[w][p] * E + l2i[w][q];
  }
  float sc[16]; int si[16];
#pragma unroll
  for (int j = 0; j < 16; ++j) {
    float lm = -1e30f; int li = 0, lp = 0;
#pragma unroll
    for (int e = 0; e < 8; ++e) if (pv[e] > lm) { lm = pv[e]; li = lane * 8 + e; lp = pp[e]; }
#pragma unroll
    for (int off = 16; off > 0; off >>= 1) {
      float ov = __shfl_xor(lm, off, 32);
      int   oi = __shfl_xor(li, off, 32);
      int   op = __shfl_xor(lp, off, 32);
      if (ov > lm || (ov == lm && oi < li)) { lm = ov; li = oi; lp = op; }
    }
    sc[j] = lm; si[j] = lp;
    if ((li >> 3) == lane) {
#pragma unroll
      for (int e = 0; e < 8; ++e) if (e == (li & 7)) pv[e] = -1e30f;
    }
  }
  if (lane == 0) {
    float mx = sc[0], sum = 0.f;
#pragma unroll
    for (int j = 0; j < 16; ++j) sum += __expf(sc[j] - mx);
    float inv = 1.f / sum;
#pragma unroll
    for (int j = 0; j < 16; ++j) {
      idxg[row * K + j] = si[j];
      wgtg[row * K + j] = __expf(sc[j] - mx) * inv;
    }
  }
}

// ---------------------------------------------------------------------------
// Heavy fused kernels. Shared geometry:
// grid(8 token-tiles of 128, 64 slot-groups of 2048), 512 threads = 16 waves.
// Per 32-slot chunk: gather wd/wu rows -> LDS bf16; phase1 C=relu(Q·Wd^T)∘w via
// WMMA; phase2 acc += C·Wu via WMMA with 128 persistent acc VGPRs per wave.
// ---------------------------------------------------------------------------
constexpr int TM = 128;      // token tile
constexpr int CH = 32;       // slots per chunk
constexpr int GROUPS = 64;
constexpr int GROUP_SLOTS = SLOTS / GROUPS;  // 2048
constexpr int NIT = GROUP_SLOTS / CH;        // 64
constexpr int BUF_BYTES = CH * D * 2;        // 32 KB per LDS tile buffer

// ---- phase-1/phase-2 compute shared by both variants ----------------------
__device__ __forceinline__ void moe_phase1(const bf16* __restrict__ Qb, const bf16* wdl,
                                           const float* wl, bf16* cs,
                                           int tok_base, int w, int lane) {
  int mt = w >> 1, nt = w & 1;
  v8f c = {};
  const bf16* Ab = Qb  + (size_t)(tok_base + mt * 16) * D;
  const bf16* Bb = wdl + (nt * 16) * D;
#pragma unroll
  for (int ks = 0; ks < D / 32; ++ks) {
    v16bf a = load_frag_a (Ab + ks * 32, D, lane);
    v16bf b = load_frag_bt(Bb + ks * 32, D, lane);
    c = wmma_bf16(a, b, c);
  }
  int n = lane & 15, h = lane >> 4;
  float wv = wl[nt * 16 + n];
#pragma unroll
  for (int r = 0; r < 8; ++r) {
    int m = mt * 16 + r + 8 * h;
    cs[m * CH + nt * 16 + n] = (bf16)(fmaxf(c[r], 0.f) * wv);
  }
}

__device__ __forceinline__ void moe_phase2(const bf16* cs, const bf16* wul,
                                           v8f acc[8][2], int w, int lane) {
  v16bf bfr[2];
#pragma unroll
  for (int ns = 0; ns < 2; ++ns)
    bfr[ns] = load_frag_b(wul + w * 32 + ns * 16, D, lane);
#pragma unroll
  for (int mt = 0; mt < 8; ++mt) {
    v16bf a = load_frag_a(cs + mt * 16 * CH, CH, lane);
#pragma unroll
    for (int ns = 0; ns < 2; ++ns)
      acc[mt][ns] = wmma_bf16(a, bfr[ns], acc[mt][ns]);
  }
}

__device__ __forceinline__ void moe_writeout(float* __restrict__ out, v8f acc[8][2],
                                             int tok_base, int w, int lane) {
  int n = lane & 15, h = lane >> 4;
#pragma unroll
  for (int mt = 0; mt < 8; ++mt)
#pragma unroll
    for (int ns = 0; ns < 2; ++ns) {
      int col = w * 32 + ns * 16 + n;
#pragma unroll
      for (int r = 0; r < 8; ++r) {
        int row = tok_base + mt * 16 + r + 8 * h;
        atomicAdd(&out[row * D + col], acc[mt][ns][r]);
      }
    }
}

// ---- variant A: f32 tables, synchronous gather + convert (fallback) -------
constexpr size_t MOE_LDS = 2 * (size_t)BUF_BYTES + (size_t)TM * CH * 2 + CH * 4; // 73856

__global__ __launch_bounds__(512)
void moe_kernel(const bf16* __restrict__ Qb, const float* __restrict__ w_down,
                const float* __restrict__ w_up, const int* __restrict__ idxg,
                const float* __restrict__ wgtg, float* __restrict__ out) {
  extern __shared__ __align__(16) char smem[];
  bf16*  wdl = (bf16*)smem;                               // CH x D
  bf16*  wul = (bf16*)(smem + BUF_BYTES);                 // CH x D
  bf16*  cs  = (bf16*)(smem + 2 * BUF_BYTES);             // TM x CH
  float* wl  = (float*)(smem + 2 * BUF_BYTES + TM * CH * 2); // CH

  int lane = threadIdx.x & 31;
  int w    = threadIdx.x >> 5;
  int tok_base = blockIdx.x * TM;
  int slot0    = blockIdx.y * GROUP_SLOTS;

  v8f acc[8][2];
#pragma unroll
  for (int mt = 0; mt < 8; ++mt)
#pragma unroll
    for (int ns = 0; ns < 2; ++ns) { v8f z = {}; acc[mt][ns] = z; }

  for (int it = 0; it < NIT; ++it) {
    int sbase = slot0 + it * CH;
    if (threadIdx.x < CH) wl[threadIdx.x] = wgtg[sbase + threadIdx.x];
    if (threadIdx.x == 0 && it + 1 < NIT)
      __builtin_prefetch(idxg + sbase + CH, 0, 1);
    {
      int r    = threadIdx.x >> 4;
      int part = threadIdx.x & 15;
      int rowd = idxg[sbase + r];
      const float4* srcd = (const float4*)(w_down + (size_t)rowd * D) + part * 8;
      const float4* srcu = (const float4*)(w_up   + (size_t)rowd * D) + part * 8;
      bf16* dd = wdl + r * D + part * 32;
      bf16* du = wul + r * D + part * 32;
#pragma unroll
      for (int i = 0; i < 8; ++i) {
        float4 a = srcd[i];
        dd[i * 4 + 0] = (bf16)a.x; dd[i * 4 + 1] = (bf16)a.y;
        dd[i * 4 + 2] = (bf16)a.z; dd[i * 4 + 3] = (bf16)a.w;
        float4 b = srcu[i];
        du[i * 4 + 0] = (bf16)b.x; du[i * 4 + 1] = (bf16)b.y;
        du[i * 4 + 2] = (bf16)b.z; du[i * 4 + 3] = (bf16)b.w;
      }
    }
    __syncthreads();
    moe_phase1(Qb, wdl, wl, cs, tok_base, w, lane);
    __syncthreads();
    moe_phase2(cs, wul, acc, w, lane);
    __syncthreads();
  }
  moe_writeout(out, acc, tok_base, w, lane);
}

// ---- variant B: bf16 tables + GLOBAL_LOAD_ASYNC_TO_LDS, double-buffered ---
constexpr size_t MOE_ASYNC_LDS = 4 * (size_t)BUF_BYTES + (size_t)TM * CH * 2 + CH * 4; // 139392

__global__ __launch_bounds__(512)
void moe_async_kernel(const bf16* __restrict__ Qb, const bf16* __restrict__ Wd16,
                      const bf16* __restrict__ Wu16, const int* __restrict__ idxg,
                      const float* __restrict__ wgtg, float* __restrict__ out) {
  extern __shared__ __align__(16) char smem[];
  // buffer layout: [wd buf0][wd buf1][wu buf0][wu buf1][cs][wl]
  bf16*  cs = (bf16*)(smem + 4 * BUF_BYTES);
  float* wl = (float*)(smem + 4 * BUF_BYTES + TM * CH * 2);

  int lane = threadIdx.x & 31;
  int w    = threadIdx.x >> 5;
  int tok_base = blockIdx.x * TM;
  int slot0    = blockIdx.y * GROUP_SLOTS;

  // each thread async-copies 64B of one wd row and 64B of one wu row (b128 x4)
  auto issue_gather = [&](int buf, int sbase) {
    bf16* wdlb = (bf16*)(smem + (size_t)buf * BUF_BYTES);
    bf16* wulb = (bf16*)(smem + (size_t)(2 + buf) * BUF_BYTES);
    int r    = threadIdx.x >> 4;   // 32 rows, 16 threads/row
    int part = threadIdx.x & 15;   // 32 bf16 (64B) per thread
    int rowd = idxg[sbase + r];
    const bf16* gd = Wd16 + (size_t)rowd * D + part * 32;
    const bf16* gu = Wu16 + (size_t)rowd * D + part * 32;
    uint32_t ld = (uint32_t)(uintptr_t)(wdlb + r * D + part * 32);
    uint32_t lu = (uint32_t)(uintptr_t)(wulb + r * D + part * 32);
#pragma unroll
    for (int i = 0; i < 4; ++i) {
      async_copy_b128(ld + i * 16, (const char*)gd + i * 16);
      async_copy_b128(lu + i * 16, (const char*)gu + i * 16);
    }
  };

  v8f acc[8][2];
#pragma unroll
  for (int mt = 0; mt < 8; ++mt)
#pragma unroll
    for (int ns = 0; ns < 2; ++ns) { v8f z = {}; acc[mt][ns] = z; }

  issue_gather(0, slot0);                 // prime the pipeline

  for (int it = 0; it < NIT; ++it) {
    int buf   = it & 1;
    int sbase = slot0 + it * CH;
    wait_async0();                        // this wave's gathered bytes are in LDS
    __syncthreads();                      // every wave's gather complete
    if (threadIdx.x < CH) wl[threadIdx.x] = wgtg[sbase + threadIdx.x];
    if (it + 1 < NIT) issue_gather(buf ^ 1, sbase + CH);   // overlap next chunk
    __syncthreads();                      // wl visible to all waves
    {
      const bf16* wdlb = (const bf16*)(smem + (size_t)buf * BUF_BYTES);
      moe_phase1(Qb, wdlb, wl, cs, tok_base, w, lane);
    }
    __syncthreads();
    {
      const bf16* wulb = (const bf16*)(smem + (size_t)(2 + buf) * BUF_BYTES);
      moe_phase2(cs, wulb, acc, w, lane);
    }
    __syncthreads();                      // done reading buf before it is refilled
  }
  moe_writeout(out, acc, tok_base, w, lane);
}

// ---------------------------------------------------------------------------
extern "C" void kernel_launch(void* const* d_in, const int* in_sizes, int n_in,
                              void* d_out, int out_size, void* d_ws, size_t ws_size,
                              hipStream_t stream) {
  const float* queries = (const float*)d_in[0];   // NT x D
  const float* Wq      = (const float*)d_in[1];   // HD x D
  const float* bq      = (const float*)d_in[2];   // HD
  const float* keys    = (const float*)d_in[3];   // H x 2 x E x HALF
  const float* w_down  = (const float*)d_in[4];   // EE x D
  const float* w_up    = (const float*)d_in[5];   // EE x D
  float* out = (float*)d_out;                     // NT x D

  char* ws = (char*)d_ws;
  bf16*  Qb   = (bf16*) (ws);                               // 1 MB
  bf16*  Wqb  = (bf16*) (ws + (1u  << 20));                 // 1 MB
  float* qh   = (float*)(ws + (2u  << 20));                 // 4 MB
  float* s1   = (float*)(ws + (6u  << 20));                 // 8 MB
  float* s2   = (float*)(ws + (14u << 20));                 // 8 MB
  int*   idxg = (int*)  (ws + (22u << 20));                 // 512 KB
  float* wgtg = (float*)(ws + (22u << 20) + (512u << 10));  // 512 KB
  bf16*  Wd16 = (bf16*) (ws + (24u << 20));                 // 64 MB (optional)
  bf16*  Wu16 = (bf16*) (ws + (88u << 20));                 // 64 MB (optional)
  const bool big_ws = ws_size >= (152ull << 20);

  cvt_bf16_kernel<<<(NT * D) / 256, 256, 0, stream>>>(queries, Qb, NT * D);
  cvt_bf16_kernel<<<(HD * D) / 256, 256, 0, stream>>>(Wq, Wqb, HD * D);
  zero_out_kernel<<<(NT * D) / 256, 256, 0, stream>>>(out, NT * D);

  qproj_kernel<<<(NT / 16) * (HD / 16) / 8, 256, 0, stream>>>(Qb, Wqb, bq, qh);

  scores_kernel<<<(2 * MROWS * E) / 256, 256, 0, stream>>>(qh, keys, s1, s2);

  topk_kernel<<<MROWS / 4, 128, 0, stream>>>(s1, s2, idxg, wgtg);

  dim3 grid(NT / TM, GROUPS);
  if (big_ws) {
    int n4 = EE * D / 4;
    cvt_tables4_kernel<<<n4 / 256, 256, 0, stream>>>(w_down, Wd16, n4);
    cvt_tables4_kernel<<<n4 / 256, 256, 0, stream>>>(w_up,   Wu16, n4);
    moe_async_kernel<<<grid, 512, MOE_ASYNC_LDS, stream>>>(Qb, Wd16, Wu16, idxg, wgtg, out);
  } else {
    moe_kernel<<<grid, 512, MOE_LDS, stream>>>(Qb, w_down, w_up, idxg, wgtg, out);
  }
}
